// CrossAttentionBlock_79809082294445
// MI455X (gfx1250) — matrix operand order repaired and checked by reading, not compiled, not run on hardware
//
#include <hip/hip_runtime.h>

// ---------------------------------------------------------------------------
// CrossAttentionBlock for MI455X (gfx1250): dsconv projections + flash attn
// f16 WMMA (16x16x32) everywhere; score matrix never hits HBM.
// LDS tiles padded for the 64-bank x 4B LDS: GEMM row stride 72 halfs (144B),
// attention P-tile row stride 40 halfs (80B) -> conflict-free ds_load_b128.
// ---------------------------------------------------------------------------

typedef _Float16 half_t;
typedef __attribute__((ext_vector_type(16))) _Float16 v16h;
typedef __attribute__((ext_vector_type(8)))  float    v8f;

#define CDIM   256
#define NHEAD  8
#define HDIM   32
#define HWN    4096   // 64*64 tokens
#define NB     2
#define SCALE_INV 0.17677669529663687f  // 1/sqrt(32)

#define LDSA_STRIDE 72   // halfs per row (64 data + 8 pad) -> 144B, bank-conflict free
#define LDSP_STRIDE 40   // halfs per row (32 data + 8 pad) -> 80B, bank-conflict free

// Load a v16h fragment from two 16-byte chunks (global or LDS).
static __device__ inline v16h ld_v16h(const half_t* lo, const half_t* hi) {
  union { v16h v; uint4 u[2]; } r;
  r.u[0] = *reinterpret_cast<const uint4*>(lo);
  r.u[1] = *reinterpret_cast<const uint4*>(hi);
  return r.v;
}

static __device__ inline v8f wmma16x16x32(v16h a, v16h b, v8f c) {
  return __builtin_amdgcn_wmma_f32_16x16x32_f16(
      /*neg_a=*/false, a, /*neg_b=*/false, b,
      /*c_mod=*/(short)0, c, /*reuse_a=*/false, /*reuse_b=*/false);
}

// ---------------------------------------------------------------------------
// Kernel 1: depthwise 3x3 (SAME, zero pad) + bias for q(x), k(ctx), v(ctx).
// Output f16, layout t[proj][b][c][p], p = h*64+w.
// ---------------------------------------------------------------------------
__global__ __launch_bounds__(256)
void ca_dwconv3x3_kernel(const float* __restrict__ x, const float* __restrict__ ctx,
                         const float* __restrict__ qdw, const float* __restrict__ qdwb,
                         const float* __restrict__ kdw, const float* __restrict__ kdwb,
                         const float* __restrict__ vdw, const float* __restrict__ vdwb,
                         half_t* __restrict__ t_h) {
  const int zb   = blockIdx.z;          // proj*2 + b
  const int proj = zb >> 1;
  const int b    = zb & 1;
  const int c    = blockIdx.y;
  const int h    = blockIdx.x * 4 + (threadIdx.x >> 6);
  const int w    = threadIdx.x & 63;

  const float* src = (proj == 0) ? x : ctx;
  const float* dwp = (proj == 0) ? qdw : ((proj == 1) ? kdw : vdw);
  const float* dwb = (proj == 0) ? qdwb : ((proj == 1) ? kdwb : vdwb);

  const float* sp = src + ((size_t)(b * CDIM + c)) * HWN;
  float wgt[9];
#pragma unroll
  for (int i = 0; i < 9; ++i) wgt[i] = dwp[c * 9 + i];

  float acc = dwb[c];
#pragma unroll
  for (int dy = 0; dy < 3; ++dy) {
#pragma unroll
    for (int dx = 0; dx < 3; ++dx) {
      const int hh = h + dy - 1;
      const int ww = w + dx - 1;
      if (hh >= 0 && hh < 64 && ww >= 0 && ww < 64)
        acc += wgt[dy * 3 + dx] * sp[hh * 64 + ww];
    }
  }
  t_h[(((size_t)proj * NB + b) * CDIM + c) * HWN + h * 64 + w] = (half_t)acc;
}

// ---------------------------------------------------------------------------
// Kernel 2: pointwise 1x1 = GEMM  Y[co,p] = sum_ci W[co,ci] * T[ci,p]  (+bias)
// per (proj, b).  Block tile 64(M) x 64(N), K staged 64-wide through LDS.
// Q/K stored [b][head][pos][d] f16; V stored [b][head][d][pos] f16.
// ---------------------------------------------------------------------------
__global__ __launch_bounds__(256)
void ca_pw_gemm_wmma_kernel(const half_t* __restrict__ t_h,
                            const float* __restrict__ qpw, const float* __restrict__ qpwb,
                            const float* __restrict__ kpw, const float* __restrict__ kpwb,
                            const float* __restrict__ vpw, const float* __restrict__ vpwb,
                            half_t* __restrict__ q_h, half_t* __restrict__ k_h,
                            half_t* __restrict__ v_h) {
  __shared__ half_t lds_a[64 * LDSA_STRIDE];   // A tile: [m][k] row-major, padded
  __shared__ half_t lds_bt[64 * LDSA_STRIDE];  // B tile transposed: [n][k], padded

  const int zb   = blockIdx.z;
  const int proj = zb >> 1;
  const int b    = zb & 1;
  const int mbase = blockIdx.y * 64;  // c_out
  const int nbase = blockIdx.x * 64;  // p

  const float* Wp = (proj == 0) ? qpw : ((proj == 1) ? kpw : vpw);
  const float* Bp = (proj == 0) ? qpwb : ((proj == 1) ? kpwb : vpwb);
  const half_t* Tp = t_h + ((size_t)proj * NB + b) * CDIM * HWN;

  const int wave   = threadIdx.x >> 5;
  const int lane   = threadIdx.x & 31;
  const int lane16 = lane & 15;
  const int hh     = lane >> 4;            // lane half (0/1)
  const int wave_m = (wave & 3) * 16;      // 4 m-tiles of 16
  const int n0     = (wave >> 2) * 32;     // 2 n-tiles of 16 per wave

  v8f acc0 = {};
  v8f acc1 = {};

  for (int kb = 0; kb < CDIM; kb += 64) {
    // --- cooperative stage: A (weights, f32 -> f16) ---
    {
      const int row  = threadIdx.x >> 2;        // 0..63  (c_out within tile)
      const int coff = (threadIdx.x & 3) << 4;  // 0,16,32,48 (k within tile)
      union { float4 f4[4]; float f[16]; } ua;
      const float4* gp =
          reinterpret_cast<const float4*>(Wp + (size_t)(mbase + row) * CDIM + kb + coff);
#pragma unroll
      for (int i = 0; i < 4; ++i) ua.f4[i] = gp[i];
#pragma unroll
      for (int i = 0; i < 16; ++i)
        lds_a[row * LDSA_STRIDE + coff + i] = (half_t)ua.f[i];

      // --- cooperative stage: B (activations f16), stored transposed ---
      const int k    = row;                     // c_in within tile
      const int noff = coff;                    // p within tile
      union { uint4 u4[2]; half_t hx[16]; } ub;
      const uint4* tp =
          reinterpret_cast<const uint4*>(Tp + (size_t)(kb + k) * HWN + nbase + noff);
      ub.u4[0] = tp[0];
      ub.u4[1] = tp[1];
#pragma unroll
      for (int i = 0; i < 16; ++i)
        lds_bt[(noff + i) * LDSA_STRIDE + k] = ub.hx[i];
    }
    __syncthreads();

#pragma unroll
    for (int ks = 0; ks < 64; ks += 32) {
      // A fragment: lane holds row M=lane16, K halfs {8hh..+7} and {16+8hh..+7}
      const half_t* ap = &lds_a[(wave_m + lane16) * LDSA_STRIDE + ks + 8 * hh];
      v16h aF = ld_v16h(ap, ap + 16);
      // B fragments: lane holds col N=lane16, K halfs {8hh..+7} and {16+8hh..+7}
      const half_t* bp0 = &lds_bt[(n0 + lane16) * LDSA_STRIDE + ks + 8 * hh];
      const half_t* bp1 = &lds_bt[(n0 + 16 + lane16) * LDSA_STRIDE + ks + 8 * hh];
      v16h bF0 = ld_v16h(bp0, bp0 + 16);
      v16h bF1 = ld_v16h(bp1, bp1 + 16);
      acc0 = wmma16x16x32(aF, bF0, acc0);
      acc1 = wmma16x16x32(aF, bF1, acc1);
    }
    __syncthreads();
  }

  // Epilogue: C layout -> lane holds (M = r + 8*hh, N = lane16)
#pragma unroll
  for (int t = 0; t < 2; ++t) {
    v8f a = (t == 0) ? acc0 : acc1;
#pragma unroll
    for (int r = 0; r < 8; ++r) {
      const int co = mbase + wave_m + r + 8 * hh;
      const int p  = nbase + n0 + 16 * t + lane16;
      const half_t hv = (half_t)(a[r] + Bp[co]);
      if (proj == 2) {
        // V: [b][c][p] == [b][head][d][pos]
        v_h[((size_t)b * CDIM + co) * HWN + p] = hv;
      } else {
        const int head = co >> 5, d = co & 31;
        half_t* dst = (proj == 0) ? q_h : k_h;
        dst[(((size_t)b * NHEAD + head) * HWN + p) * HDIM + d] = hv;
      }
    }
  }
}

// ---------------------------------------------------------------------------
// Kernel 3: flash attention.  One wave per 16-query tile of one (b, head).
// Keys streamed in 32-wide chunks: 2 score WMMAs, online softmax, P transposed
// through per-wave LDS, 2 PV WMMAs (hd=32 -> two 16-wide N tiles).
// ---------------------------------------------------------------------------
__global__ __launch_bounds__(256)
void ca_flash_attn_wmma_kernel(const half_t* __restrict__ q_h,
                               const half_t* __restrict__ k_h,
                               const half_t* __restrict__ v_h,
                               float* __restrict__ out) {
  __shared__ half_t lds_p[8 * 16 * LDSP_STRIDE];  // per-wave 16-row P tile, padded

  const int wave   = threadIdx.x >> 5;
  const int lane   = threadIdx.x & 31;
  const int lane16 = lane & 15;
  const int hh     = lane >> 4;

  const int task  = blockIdx.x * 8 + wave;        // 0 .. 2*8*256-1
  const int b     = task >> 11;
  const int head  = (task >> 8) & 7;
  const int qbase = (task & 255) * 16;

  const half_t* qh = q_h + ((size_t)b * NHEAD + head) * HWN * HDIM;
  const half_t* kh = k_h + ((size_t)b * NHEAD + head) * HWN * HDIM;
  const half_t* vh = v_h + ((size_t)b * CDIM + head * HDIM) * HWN;
  half_t* pbase = lds_p + wave * (16 * LDSP_STRIDE);

  // Q A-fragment (16x32, K = head dim): row = qbase + lane16
  const half_t* qp = qh + (size_t)(qbase + lane16) * HDIM + 8 * hh;
  const v16h qF = ld_v16h(qp, qp + 16);

  float m_run[8], l_run[8];
#pragma unroll
  for (int r = 0; r < 8; ++r) { m_run[r] = -3.0e38f; l_run[r] = 0.0f; }
  v8f o0 = {};
  v8f o1 = {};
  const v8f zero = {};

  for (int kc = 0; kc < HWN / 32; ++kc) {
    // prefetch next chunk of K and V rows
    const int nkc = (kc + 1) & (HWN / 32 - 1);
    __builtin_prefetch(kh + (size_t)(nkc * 32 + lane16) * HDIM, 0, 3);
    __builtin_prefetch(vh + (size_t)lane16 * HWN + nkc * 32, 0, 3);

    // K^T B-fragments: lane holds col = key, d {8hh..+7} and {16+8hh..+7}
    const half_t* kp0 = kh + (size_t)(kc * 32 + lane16) * HDIM + 8 * hh;
    const half_t* kp1 = kh + (size_t)(kc * 32 + 16 + lane16) * HDIM + 8 * hh;
    v16h kF0 = ld_v16h(kp0, kp0 + 16);
    v16h kF1 = ld_v16h(kp1, kp1 + 16);

    v8f s0 = wmma16x16x32(qF, kF0, zero);
    v8f s1 = wmma16x16x32(qF, kF1, zero);

    // online softmax per row (row M = r + 8*hh; cols striped across 16 lanes)
#pragma unroll
    for (int r = 0; r < 8; ++r) {
      float a = s0[r] * SCALE_INV;
      float c = s1[r] * SCALE_INV;
      float mx = fmaxf(a, c);
#pragma unroll
      for (int mk = 1; mk < 16; mk <<= 1) mx = fmaxf(mx, __shfl_xor(mx, mk, 32));
      const float mnew = fmaxf(m_run[r], mx);
      const float corr = __expf(m_run[r] - mnew);
      m_run[r] = mnew;
      const float p0 = __expf(a - mnew);
      const float p1 = __expf(c - mnew);
      float ps = p0 + p1;
#pragma unroll
      for (int mk = 1; mk < 16; mk <<= 1) ps += __shfl_xor(ps, mk, 32);
      l_run[r] = l_run[r] * corr + ps;
      o0[r] *= corr;
      o1[r] *= corr;
      // stash P (C layout -> row-major LDS tile) for the transpose
      const int mrow = r + 8 * hh;
      pbase[mrow * LDSP_STRIDE + lane16]      = (half_t)p0;
      pbase[mrow * LDSP_STRIDE + 16 + lane16] = (half_t)p1;
    }

    // same-wave cross-lane LDS RAW: order ds_load after ds_store
    asm volatile("s_wait_dscnt 0x0" ::: "memory");

    // P A-fragment (16 q x 32 keys): row = lane16
    const half_t* pp = pbase + lane16 * LDSP_STRIDE + 8 * hh;
    v16h pF = ld_v16h(pp, pp + 16);

    // V B-fragments: lane holds col d = 16t + lane16, keys {8hh..+7} and {16+8hh..+7}
    const half_t* vp0 = vh + (size_t)lane16 * HWN + kc * 32 + 8 * hh;
    const half_t* vp1 = vh + (size_t)(16 + lane16) * HWN + kc * 32 + 8 * hh;
    v16h vF0 = ld_v16h(vp0, vp0 + 16);
    v16h vF1 = ld_v16h(vp1, vp1 + 16);

    o0 = wmma16x16x32(pF, vF0, o0);
    o1 = wmma16x16x32(pF, vF1, o1);

    asm volatile("s_wait_dscnt 0x0" ::: "memory");  // stores of next iter vs loads
  }

  // Epilogue: out[b][head*32 + d][qbase + m], m = r + 8*hh contiguous over r.
  float e[8];
#pragma unroll
  for (int t = 0; t < 2; ++t) {
    v8f o = (t == 0) ? o0 : o1;
#pragma unroll
    for (int r = 0; r < 8; ++r) e[r] = o[r] / l_run[r];
    const size_t base =
        ((size_t)b * CDIM + head * HDIM + 16 * t + lane16) * HWN + qbase + 8 * hh;
    float4 lo4 = {e[0], e[1], e[2], e[3]};
    float4 hi4 = {e[4], e[5], e[6], e[7]};
    *reinterpret_cast<float4*>(out + base)     = lo4;
    *reinterpret_cast<float4*>(out + base + 4) = hi4;
  }
}

// ---------------------------------------------------------------------------
// Host launcher
// ---------------------------------------------------------------------------
extern "C" void kernel_launch(void* const* d_in, const int* in_sizes, int n_in,
                              void* d_out, int out_size, void* d_ws, size_t ws_size,
                              hipStream_t stream) {
  const float* x    = (const float*)d_in[0];
  const float* ctx  = (const float*)d_in[1];
  const float* qdw  = (const float*)d_in[2];
  const float* qdwb = (const float*)d_in[3];
  const float* qpw  = (const float*)d_in[4];
  const float* qpwb = (const float*)d_in[5];
  const float* kdw  = (const float*)d_in[6];
  const float* kdwb = (const float*)d_in[7];
  const float* kpw  = (const float*)d_in[8];
  const float* kpwb = (const float*)d_in[9];
  const float* vdw  = (const float*)d_in[10];
  const float* vdwb = (const float*)d_in[11];
  const float* vpw  = (const float*)d_in[12];
  const float* vpwb = (const float*)d_in[13];
  float* out = (float*)d_out;

  const size_t n_cp = (size_t)NB * CDIM * HWN;  // per-array element count
  half_t* t_h = (half_t*)d_ws;                  // 3 * n_cp (dwconv outputs, f16)
  half_t* q_h = t_h + 3 * n_cp;                 // [b][head][pos][d]
  half_t* k_h = q_h + n_cp;                     // [b][head][pos][d]
  half_t* v_h = k_h + n_cp;                     // [b][head][d][pos]

  // 1) depthwise 3x3 + bias for the three projections
  ca_dwconv3x3_kernel<<<dim3(16, CDIM, 6), 256, 0, stream>>>(
      x, ctx, qdw, qdwb, kdw, kdwb, vdw, vdwb, t_h);

  // 2) pointwise 1x1 as WMMA GEMM (per proj, per batch)
  ca_pw_gemm_wmma_kernel<<<dim3(HWN / 64, CDIM / 64, 6), 256, 0, stream>>>(
      t_h, qpw, qpwb, kpw, kpwb, vpw, vpwb, q_h, k_h, v_h);

  // 3) flash attention (one wave per 16-query tile)
  ca_flash_attn_wmma_kernel<<<dim3((NB * NHEAD * (HWN / 16)) / 8), 256, 0, stream>>>(
      q_h, k_h, v_h, out);
}
